// SparseGATRefinement_69784628625715
// MI455X (gfx1250) — matrix operand order repaired and checked by reading, not compiled, not run on hardware
//
#include <hip/hip_runtime.h>
#include <hip/hip_bf16.h>

// MI455X / gfx1250, wave32. Sparse GAT refinement pipeline.
// GEMMs use v_wmma_f32_16x16x32_f16 (f32 accum). Everything else VALU/LDS.

typedef __attribute__((ext_vector_type(16))) _Float16 v16h;
typedef __attribute__((ext_vector_type(8)))  float    v8f;

#define NNODES 4096
#define KNN    16
#define DEG    (KNN + 1)       // + self loop
#define NB     4096            // histogram bins over (0.3, 1.0]
#define NPIX   65536           // 256*256
#define NBATCH 4
#define FCH    64
#define THRV   0.3f

// ---------------------------------------------------------------- utilities
__global__ void sgat_copy4(const float4* __restrict__ s, float4* __restrict__ d, long n) {
  long i = (long)blockIdx.x * blockDim.x + threadIdx.x;
  long stride = (long)gridDim.x * blockDim.x;
  for (; i < n; i += stride) d[i] = s[i];
}

__global__ void sgat_zero_i32(int* p, int n) {
  int i = blockIdx.x * blockDim.x + threadIdx.x;
  int stride = gridDim.x * blockDim.x;
  for (; i < n; i += stride) p[i] = 0;
}

__global__ void sgat_fill_i32(int* p, int n, int v) {
  int i = blockIdx.x * blockDim.x + threadIdx.x;
  int stride = gridDim.x * blockDim.x;
  for (; i < n; i += stride) p[i] = v;
}

__global__ void sgat_nodes_init(int* nodes) {       // safety pad: valid pixel ids
  int g = blockIdx.x * blockDim.x + threadIdx.x;    // NBATCH*NNODES
  nodes[g] = g & (NNODES - 1);
}

// ------------------------------------------------- top-4096 selection (radix)
__global__ void sgat_hist(const float* __restrict__ prob, int* __restrict__ hist) {
  int g = blockIdx.x * blockDim.x + threadIdx.x;    // NBATCH*NPIX
  int b = g >> 16;
  float p = prob[g];
  if (p > THRV) {
    int bin = (int)((p - THRV) * ((float)NB / 0.7f));
    if (bin > NB - 1) bin = NB - 1;
    atomicAdd(&hist[b * NB + bin], 1);
  }
}

__global__ void sgat_cutoff(const int* __restrict__ hist, int* __restrict__ meta) {
  int b = blockIdx.x;
  if (threadIdx.x != 0) return;
  int acc = 0, cutoff = -1;
  for (int bin = NB - 1; bin >= 0; --bin) {
    int c = hist[b * NB + bin];
    if (acc + c >= NNODES) { cutoff = bin; break; }
    acc += c;
  }
  meta[b * 4 + 0] = cutoff;           // bin at the selection boundary (-1: take all)
  meta[b * 4 + 1] = acc;              // strictly-above count
  meta[b * 4 + 2] = NNODES - acc;     // how many to take from cutoff bin
}

__global__ void sgat_compact(const float* __restrict__ prob, const int* __restrict__ meta,
                             int* __restrict__ cnt, int* __restrict__ nodes) {
  int g = blockIdx.x * blockDim.x + threadIdx.x;    // NBATCH*NPIX
  int b = g >> 16, pix = g & (NPIX - 1);
  float p = prob[g];
  if (p <= THRV) return;
  int bin = (int)((p - THRV) * ((float)NB / 0.7f));
  if (bin > NB - 1) bin = NB - 1;
  int cutoff = meta[b * 4 + 0], n_above = meta[b * 4 + 1], n_fill = meta[b * 4 + 2];
  if (bin > cutoff) {
    int pos = atomicAdd(&cnt[b * 2 + 0], 1);
    if (pos < NNODES) nodes[b * NNODES + pos] = pix;
  } else if (bin == cutoff) {
    int pos = atomicAdd(&cnt[b * 2 + 1], 1);
    if (pos < n_fill) nodes[b * NNODES + n_above + pos] = pix;
  }
}

// ------------------------------------------------------------- node features
__global__ void sgat_gather(const float* __restrict__ feat, const int* __restrict__ nodes,
                            float* __restrict__ nf) {
  int b = blockIdx.y;
  int nl = threadIdx.x >> 6, c = threadIdx.x & 63;
  int n = blockIdx.x * 4 + nl;                      // 1024 blocks * 4 nodes
  int pix = nodes[b * NNODES + n];
  nf[((size_t)(b * NNODES + n)) * FCH + c] = feat[((size_t)(b * FCH + c)) * NPIX + pix];
}

// ------------------------------------------------------ WMMA GEMM  Y = X*W^T+b
// X: (nrows,64) f32, W: (64,64) row-major (we need W^T), Y: (nrows,64) f32.
// Block = 128 threads = 4 waves; wave w owns output cols [16w,16w+16).
// K=64 -> two 16x16x32 f16 WMMA steps with f32 accumulation.
__global__ void sgat_gemm64(const float* __restrict__ X, const float* __restrict__ W,
                            const float* __restrict__ bias, float* __restrict__ Y,
                            int nrows, int relu) {
  int lane = threadIdx.x & 31;
  int wave = threadIdx.x >> 5;
  int m0 = blockIdx.x * 16;
  if (m0 >= nrows) return;                           // uniform per block
  int n0 = wave * 16;
  int half = lane >> 4;                              // 0: lanes 0-15, 1: 16-31
  int l15 = lane & 15;

  v8f acc = {};
#pragma unroll
  for (int ks = 0; ks < 2; ++ks) {
    v16h a, bf;
    // A fragment: 16x32 f16. lane<16: row=l15, K=0..15;
    // lane>=16: same row, K=16..31.
    const float* xrow = X + (size_t)(m0 + l15) * FCH + ks * 32;
#pragma unroll
    for (int j = 0; j < 16; ++j) {
      int k = j + half * 16;
      a[j] = (_Float16)xrow[k];
    }
    // B fragment: 32x16 f16, B[k][n] = W[n][k]. lanes 0-15: col=n, K=0..15;
    // lanes 16-31: col=n, K=16..31.
    const float* wrow = W + (size_t)(n0 + l15) * FCH + ks * 32;
#pragma unroll
    for (int j = 0; j < 16; ++j) {
      int k = j + half * 16;
      bf[j] = (_Float16)wrow[k];
    }
    acc = __builtin_amdgcn_wmma_f32_16x16x32_f16(false, a, false, bf,
                                                 (short)0, acc, false, false);
  }
  // C/D layout: VGPR r -> M = r (lanes 0-15) / r+8 (lanes 16-31); N = lane&15.
  int col = n0 + l15;
  float bv = bias[col];
#pragma unroll
  for (int r = 0; r < 8; ++r) {
    int row = m0 + r + half * 8;
    float v = acc[r] + bv;
    if (relu) v = fmaxf(v, 0.f);
    Y[(size_t)row * FCH + col] = v;
  }
}

// ----------------------------------------------------------------- brute KNN
__global__ void sgat_knn(const int* __restrict__ nodes, int* __restrict__ nbr) {
  int b = blockIdx.y;
  int q = blockIdx.x * blockDim.x + threadIdx.x;     // 16*256 = 4096 queries
  const int* nb = nodes + b * NNODES;
  int mypix = nb[q];
  float qy = (float)(mypix >> 8) * (1.f / 256.f);
  float qx = (float)(mypix & 255) * (1.f / 256.f);

  float d[DEG];
  int   id[DEG];
#pragma unroll
  for (int i = 0; i < DEG; ++i) { d[i] = 3.4e38f; id[i] = 0x7fffffff; }

  __shared__ float2 sc[256];
  for (int t = 0; t < NNODES; t += 256) {
    __syncthreads();
    int cp = nb[t + threadIdx.x];
    sc[threadIdx.x] = make_float2((float)(cp >> 8) * (1.f / 256.f),
                                  (float)(cp & 255) * (1.f / 256.f));
    __syncthreads();
    for (int j = 0; j < 256; ++j) {
      float dy = qy - sc[j].x, dx = qx - sc[j].y;
      float dd = dy * dy + dx * dx;
      int ci = t + j;
      if (dd < d[DEG - 1] || (dd == d[DEG - 1] && ci < id[DEG - 1])) {
        d[DEG - 1] = dd; id[DEG - 1] = ci;
#pragma unroll
        for (int s = DEG - 1; s > 0; --s) {
          bool sw = (d[s] < d[s - 1]) || (d[s] == d[s - 1] && id[s] < id[s - 1]);
          if (sw) {
            float td = d[s]; d[s] = d[s - 1]; d[s - 1] = td;
            int   ti = id[s]; id[s] = id[s - 1]; id[s - 1] = ti;
          }
        }
      }
    }
  }
  // position 0 is self (d2 == 0, unique coords) -> neighbors are 1..16
#pragma unroll
  for (int k = 0; k < KNN; ++k)
    nbr[(size_t)(b * NNODES + q) * KNN + k] = id[k + 1];
}

// ----------------------------------------------------- CSR of incoming edges
__global__ void sgat_indeg_count(const int* __restrict__ nbr, int* __restrict__ indeg) {
  int e = blockIdx.x * blockDim.x + threadIdx.x;     // NBATCH*NNODES*KNN
  int b = e / (NNODES * KNN);
  atomicAdd(&indeg[b * NNODES + nbr[e]], 1);
}

__global__ void sgat_scan(const int* __restrict__ indeg, int* __restrict__ rowst,
                          int* __restrict__ cursor) {
  int b = blockIdx.x, t = threadIdx.x;               // block 1024, 4 elems/thread
  __shared__ int part[1024];
  int base = b * NNODES + t * 4;
  int v0 = indeg[base], v1 = indeg[base + 1], v2 = indeg[base + 2], v3 = indeg[base + 3];
  int sum = v0 + v1 + v2 + v3;
  part[t] = sum;
  __syncthreads();
  for (int offs = 1; offs < 1024; offs <<= 1) {
    int x = (t >= offs) ? part[t - offs] : 0;
    __syncthreads();
    part[t] += x;
    __syncthreads();
  }
  int excl = part[t] - sum;
  int o = b * (NNODES + 1) + t * 4;
  int c = b * NNODES + t * 4;
  rowst[o + 0] = excl;               cursor[c + 0] = excl;
  rowst[o + 1] = excl + v0;          cursor[c + 1] = excl + v0;
  rowst[o + 2] = excl + v0 + v1;     cursor[c + 2] = excl + v0 + v1;
  rowst[o + 3] = excl + v0 + v1 + v2; cursor[c + 3] = excl + v0 + v1 + v2;
  if (t == 1023) rowst[b * (NNODES + 1) + NNODES] = part[1023];
}

__global__ void sgat_csr_fill(const int* __restrict__ nbr, int* __restrict__ cursor,
                              int* __restrict__ csr) {
  int g = blockIdx.x * blockDim.x + threadIdx.x;     // NBATCH*NNODES
  int b = g / NNODES, i = g % NNODES;
  int cbase = b * NNODES * DEG;
  int pos = atomicAdd(&cursor[b * NNODES + i], 1);   // self loop
  csr[cbase + pos] = i;
  for (int k = 0; k < KNN; ++k) {
    int j = nbr[(size_t)g * KNN + k];
    pos = atomicAdd(&cursor[b * NNODES + j], 1);
    csr[cbase + pos] = i;
  }
}

// ------------------------------------------------------------------ LayerNorm
__global__ void sgat_ln(const float* __restrict__ X, float* __restrict__ H,
                        const float* __restrict__ g, const float* __restrict__ bta) {
  int node = blockIdx.x * (blockDim.x >> 5) + (threadIdx.x >> 5);
  int lane = threadIdx.x & 31;
  const float* xr = X + (size_t)node * FCH;
  float v0 = xr[lane], v1 = xr[lane + 32];
  float s = v0 + v1, s2 = v0 * v0 + v1 * v1;
#pragma unroll
  for (int m = 1; m < 32; m <<= 1) { s += __shfl_xor(s, m, 32); s2 += __shfl_xor(s2, m, 32); }
  float mu = s * (1.f / 64.f);
  float var = s2 * (1.f / 64.f) - mu * mu;
  float inv = rsqrtf(var + 1e-5f);
  H[(size_t)node * FCH + lane]      = (v0 - mu) * inv * g[lane]      + bta[lane];
  H[(size_t)node * FCH + lane + 32] = (v1 - mu) * inv * g[lane + 32] + bta[lane + 32];
}

// ---------------------------------------- GATv2 aggregation (wave per node,
// online softmax over incoming edges, residual + ELU fused)
__global__ void sgat_aggregate(const float* __restrict__ xl, const float* __restrict__ xr,
                               const float* __restrict__ att, const float* __restrict__ bg,
                               const int* __restrict__ rowst, const int* __restrict__ csr,
                               float* __restrict__ x) {
  int wv = blockIdx.x * (blockDim.x >> 5) + (threadIdx.x >> 5);  // NBATCH*NNODES waves
  int b = wv >> 12, j = wv & (NNODES - 1);
  int lane = threadIdx.x & 31;
  int c0 = lane, c1 = lane + 32;                     // head(c0)=c0/16, head(c1)=c1/16
  size_t node = (size_t)(b * NNODES + j);
  float xr0 = xr[node * FCH + c0], xr1 = xr[node * FCH + c1];
  float a0 = att[c0], a1 = att[c1];
  float m0 = -3.4e38f, m1 = -3.4e38f, s0 = 0.f, s1 = 0.f, acc0 = 0.f, acc1 = 0.f;
  int e0 = rowst[b * (NNODES + 1) + j], e1 = rowst[b * (NNODES + 1) + j + 1];
  const int* csrb = csr + b * NNODES * DEG;
  for (int e = e0; e < e1; ++e) {
    size_t ni = (size_t)(b * NNODES + csrb[e]);
    float xl0 = xl[ni * FCH + c0], xl1 = xl[ni * FCH + c1];
    float t0 = xl0 + xr0; t0 = t0 > 0.f ? t0 : 0.2f * t0;   // leaky_relu(., 0.2)
    float t1 = xl1 + xr1; t1 = t1 > 0.f ? t1 : 0.2f * t1;
    float p0 = t0 * a0, p1 = t1 * a1;
#pragma unroll
    for (int m = 1; m < 16; m <<= 1) { p0 += __shfl_xor(p0, m, 32); p1 += __shfl_xor(p1, m, 32); }
    float nm0 = fmaxf(m0, p0), nm1 = fmaxf(m1, p1);
    float sc0 = __expf(m0 - nm0), sc1 = __expf(m1 - nm1);
    float w0 = __expf(p0 - nm0), w1 = __expf(p1 - nm1);
    s0 = s0 * sc0 + w0;          s1 = s1 * sc1 + w1;
    acc0 = acc0 * sc0 + w0 * xl0; acc1 = acc1 * sc1 + w1 * xl1;
    m0 = nm0; m1 = nm1;
  }
  float o0 = acc0 / s0 + bg[c0];
  float o1 = acc1 / s1 + bg[c1];
  float v0 = x[node * FCH + c0] + o0;
  float v1 = x[node * FCH + c1] + o1;
  v0 = v0 > 0.f ? v0 : __expf(v0) - 1.f;             // elu
  v1 = v1 > 0.f ? v1 : __expf(v1) - 1.f;
  x[node * FCH + c0] = v0;
  x[node * FCH + c1] = v1;
}

// ------------------------------------------------------------------ scatter
__global__ void sgat_scatter(const int* __restrict__ nodes, const float* __restrict__ y,
                             const float* __restrict__ gate, float* __restrict__ out) {
  int g = blockIdx.x * blockDim.x + threadIdx.x;     // NBATCH*NNODES*FCH
  int c = g & 63, n = (g >> 6) & (NNODES - 1), b = g >> 18;
  int pix = nodes[b * NNODES + n];
  float gs = 1.f / (1.f + __expf(-gate[0]));
  out[((size_t)(b * FCH + c)) * NPIX + pix] += gs * y[(size_t)g];
}

// ================================================================== launcher
extern "C" void kernel_launch(void* const* d_in, const int* in_sizes, int n_in,
                              void* d_out, int out_size, void* d_ws, size_t ws_size,
                              hipStream_t stream) {
  const float* feat = (const float*)d_in[0];
  const float* prob = (const float*)d_in[1];
  const float* Wp   = (const float*)d_in[2];
  const float* bp   = (const float*)d_in[3];
  const float* ln_g = (const float*)d_in[4];
  const float* ln_b = (const float*)d_in[5];
  const float* Wl   = (const float*)d_in[6];
  const float* bl   = (const float*)d_in[7];
  const float* Wr   = (const float*)d_in[8];
  const float* br   = (const float*)d_in[9];
  const float* att  = (const float*)d_in[10];
  const float* bg   = (const float*)d_in[11];
  const float* Wo   = (const float*)d_in[12];
  const float* bo   = (const float*)d_in[13];
  const float* gate = (const float*)d_in[14];
  float* out = (float*)d_out;
  (void)in_sizes; (void)n_in; (void)out_size; (void)ws_size;

  // workspace carve (all regions re-initialized or fully overwritten per call)
  char* base = (char*)d_ws; size_t off = 0;
  auto carve = [&](size_t bytes) -> void* {
    void* p = base + off; off = (off + bytes + 255) & ~(size_t)255; return p;
  };
  int* nodes  = (int*)carve((size_t)NBATCH * NNODES * 4);
  int* nbr    = (int*)carve((size_t)NBATCH * NNODES * KNN * 4);
  int* indeg  = (int*)carve((size_t)NBATCH * NNODES * 4);
  int* rowst  = (int*)carve((size_t)NBATCH * (NNODES + 1) * 4);
  int* cursor = (int*)carve((size_t)NBATCH * NNODES * 4);
  int* csr    = (int*)carve((size_t)NBATCH * NNODES * DEG * 4);
  int* hist   = (int*)carve((size_t)NBATCH * NB * 4);
  int* meta   = (int*)carve((size_t)NBATCH * 4 * 4);
  int* cnt    = (int*)carve((size_t)NBATCH * 2 * 4);
  float* xbuf = (float*)carve((size_t)NBATCH * NNODES * FCH * 4);
  float* hbuf = (float*)carve((size_t)NBATCH * NNODES * FCH * 4);
  float* xlb  = (float*)carve((size_t)NBATCH * NNODES * FCH * 4);
  float* xrb  = (float*)carve((size_t)NBATCH * NNODES * FCH * 4);

  const int NROWS = NBATCH * NNODES;                 // 16384

  // 0) out = feat (HBM-bound: ~134 MB @ 23.3 TB/s)
  sgat_copy4<<<4096, 256, 0, stream>>>((const float4*)feat, (float4*)out,
                                       (long)NBATCH * FCH * NPIX / 4);
  // 1) select top-4096 pixels per batch
  sgat_zero_i32<<<64, 256, 0, stream>>>(hist, NBATCH * NB);
  sgat_zero_i32<<<1, 64, 0, stream>>>(cnt, NBATCH * 2);
  sgat_nodes_init<<<NROWS / 256, 256, 0, stream>>>(nodes);
  sgat_hist<<<NBATCH * NPIX / 256, 256, 0, stream>>>(prob, hist);
  sgat_cutoff<<<NBATCH, 32, 0, stream>>>(hist, meta);
  sgat_compact<<<NBATCH * NPIX / 256, 256, 0, stream>>>(prob, meta, cnt, nodes);
  // 2) gather + input projection (WMMA)
  sgat_gather<<<dim3(NNODES / 4, NBATCH), 256, 0, stream>>>(feat, nodes, hbuf);
  sgat_gemm64<<<NROWS / 16, 128, 0, stream>>>(hbuf, Wp, bp, xbuf, NROWS, 0);
  // 3) brute-force 16-NN (VALU hotspot, LDS-tiled)
  sgat_knn<<<dim3(NNODES / 256, NBATCH), 256, 0, stream>>>(nodes, nbr);
  // 4) CSR of incoming edges (self loop => indeg init 1)
  sgat_fill_i32<<<64, 256, 0, stream>>>(indeg, NROWS, 1);
  sgat_indeg_count<<<NROWS * KNN / 256, 256, 0, stream>>>(nbr, indeg);
  sgat_scan<<<NBATCH, 1024, 0, stream>>>(indeg, rowst, cursor);
  sgat_csr_fill<<<NROWS / 256, 256, 0, stream>>>(nbr, cursor, csr);
  // 5) two GATv2 layers
  for (int l = 0; l < 2; ++l) {
    sgat_ln<<<NROWS / 8, 256, 0, stream>>>(xbuf, hbuf, ln_g + l * 64, ln_b + l * 64);
    sgat_gemm64<<<NROWS / 16, 128, 0, stream>>>(hbuf, Wl + l * 4096, bl + l * 64, xlb, NROWS, 0);
    sgat_gemm64<<<NROWS / 16, 128, 0, stream>>>(hbuf, Wr + l * 4096, br + l * 64, xrb, NROWS, 0);
    sgat_aggregate<<<NROWS / 8, 256, 0, stream>>>(xlb, xrb, att + l * 64, bg + l * 64,
                                                  rowst, csr, xbuf);
  }
  // 6) output head + gated scatter-add
  sgat_gemm64<<<NROWS / 16, 128, 0, stream>>>(xbuf, Wo, bo, hbuf, NROWS, 1);
  sgat_scatter<<<NROWS * FCH / 256, 256, 0, stream>>>(nodes, hbuf, gate, out);
}
